// Head_38165079392388
// MI455X (gfx1250) — matrix operand order repaired and checked by reading, not compiled, not run on hardware
//
#include <hip/hip_runtime.h>

typedef _Float16 h16v __attribute__((ext_vector_type(16)));
typedef _Float16 h8v  __attribute__((ext_vector_type(8)));
typedef float    f8v  __attribute__((ext_vector_type(8)));

#define B_SZ  512
#define T_SZ  256
#define C_SZ  384
#define HS    48
#define HSP   64            // HS padded to WMMA k multiple (zeros in 48..63)
#define VT_ST 264           // Vt row stride (halfs), padded vs bank conflicts
#define P_ST  264           // P strip row stride (halfs)
#define NWAVE 8
#define SCALE 0.14433756729740643f   // 1/sqrt(48)

union F16frag { h16v v; h8v h[2]; };

__device__ __forceinline__ f8v zero8() {
    f8v z;
#pragma unroll
    for (int i = 0; i < 8; ++i) z[i] = 0.0f;
    return z;
}

// ---- prep: W [C][HS] f32 -> wt [3][HS][C] f16 (transposed, WMMA-B friendly) ----
__global__ void head_prep_w(const float* __restrict__ Wq, const float* __restrict__ Wk,
                            const float* __restrict__ Wv, _Float16* __restrict__ wt) {
    int idx = blockIdx.x * 256 + threadIdx.x;
    const int total = 3 * HS * C_SZ;
    if (idx >= total) return;
    int m = idx / (HS * C_SZ);
    int r = idx % (HS * C_SZ);
    int n = r / C_SZ;
    int k = r % C_SZ;
    const float* W = (m == 0) ? Wq : ((m == 1) ? Wk : Wv);
    wt[idx] = (_Float16)W[k * HS + n];
}

// ---- fused causal attention: one workgroup per batch element ----
__launch_bounds__(256, 1)
__global__ void head_attn(const float* __restrict__ x, const _Float16* __restrict__ wt,
                          float* __restrict__ out) {
    __shared__ _Float16 sQ[T_SZ][HSP];          // 32 KB
    __shared__ _Float16 sK[T_SZ][HSP];          // 32 KB
    __shared__ _Float16 sVt[HS][VT_ST];         // ~25 KB, V transposed [n][k]
    __shared__ _Float16 sP[NWAVE][16][P_ST];    // ~66 KB, per-wave prob strip

    const int  b    = blockIdx.x;
    const int  tid  = threadIdx.x;
    const int  wave = tid >> 5;
    const int  lane = tid & 31;
    const int  ln16 = lane & 15;
    const bool hi   = lane >= 16;

    // zero the k-padding (cols 48..63) of Q and K; one row per thread
    {
        uint4 z = {0u, 0u, 0u, 0u};
        *(uint4*)&sQ[tid][HS]     = z;
        *(uint4*)&sQ[tid][HS + 8] = z;
        *(uint4*)&sK[tid][HS]     = z;
        *(uint4*)&sK[tid][HS + 8] = z;
    }

    // ================= Phase 1: Q,K,V = x @ W (WMMA f16 -> f32) =================
    const float* xb = x + (size_t)b * T_SZ * C_SZ;

    for (int rb = wave; rb < 16; rb += NWAVE) {
        const int m0 = rb * 16;
        f8v acc[9];
#pragma unroll
        for (int i = 0; i < 9; ++i) acc[i] = zero8();

        const float* xrow = xb + (size_t)(m0 + ln16) * C_SZ;

        for (int ks = 0; ks < 12; ++ks) {          // C = 384 = 12 * 32
            const int kb  = ks * 32;
            const int ko1 = kb + (hi ? 8 : 0);
            const int ko2 = kb + 16 + (hi ? 8 : 0);
            h16v a;
#pragma unroll
            for (int i = 0; i < 8; ++i) a[i]     = (_Float16)xrow[ko1 + i];
#pragma unroll
            for (int i = 0; i < 8; ++i) a[8 + i] = (_Float16)xrow[ko2 + i];

#pragma unroll
            for (int mt = 0; mt < 9; ++mt) {       // 3 matrices x 3 n-tiles
                const int mat = mt / 3, t = mt % 3;
                const _Float16* wrow = wt + (size_t)mat * HS * C_SZ
                                          + (size_t)(t * 16 + ln16) * C_SZ
                                          + kb + (hi ? 8 : 0);
                F16frag bf;
                bf.h[0] = *(const h8v*)(wrow);
                bf.h[1] = *(const h8v*)(wrow + 16);
                acc[mt] = __builtin_amdgcn_wmma_f32_16x16x32_f16(
                    false, a, false, bf.v, (short)0, acc[mt], false, false);
            }
        }
        // scatter C-layout results: lane holds (m = v + hi*8, n = ln16) per n-tile
#pragma unroll
        for (int t = 0; t < 3; ++t) {
#pragma unroll
            for (int v = 0; v < 8; ++v) {
                const int m = m0 + v + (hi ? 8 : 0);
                sQ[m][t * 16 + ln16]  = (_Float16)acc[0 * 3 + t][v];
                sK[m][t * 16 + ln16]  = (_Float16)acc[1 * 3 + t][v];
                sVt[t * 16 + ln16][m] = (_Float16)acc[2 * 3 + t][v];   // transposed
            }
        }
    }
    __syncthreads();

    // ============ Phase 2: S = Q K^T, causal softmax, O = P V ============
    for (int rb = wave; rb < 16; rb += NWAVE) {
        const int m0 = rb * 16;

        // Q A-fragments for k = 0..63 (two k-steps), reused across all column tiles
        F16frag qa[2];
#pragma unroll
        for (int ks = 0; ks < 2; ++ks) {
            const int kb = ks * 32;
            qa[ks].h[0] = *(const h8v*)&sQ[m0 + ln16][kb + (hi ? 8 : 0)];
            qa[ks].h[1] = *(const h8v*)&sQ[m0 + ln16][kb + 16 + (hi ? 8 : 0)];
        }

        f8v s[16];
#pragma unroll
        for (int j = 0; j < 16; ++j) {
            if (j <= rb) {
                const int n0 = j * 16;
                f8v c = zero8();
#pragma unroll
                for (int ks = 0; ks < 2; ++ks) {
                    const int kb2 = ks * 32 + (hi ? 8 : 0);
                    F16frag kf;
                    kf.h[0] = *(const h8v*)&sK[n0 + ln16][kb2];
                    kf.h[1] = *(const h8v*)&sK[n0 + ln16][kb2 + 16];
                    c = __builtin_amdgcn_wmma_f32_16x16x32_f16(
                        false, qa[ks].v, false, kf.v, (short)0, c, false, false);
                }
                if (j == rb) {                      // diagonal tile: triangular mask
#pragma unroll
                    for (int v = 0; v < 8; ++v) {
                        const int m = v + (hi ? 8 : 0);
                        s[j][v] = (ln16 <= m) ? c[v] * SCALE : -INFINITY;
                    }
                } else {
#pragma unroll
                    for (int v = 0; v < 8; ++v) s[j][v] = c[v] * SCALE;
                }
            } else {
#pragma unroll
                for (int v = 0; v < 8; ++v) s[j][v] = -INFINITY;
            }
        }

        // row softmax: C layout -> row lives in one 16-lane half, one VGPR per tile
        float rinv[8];
#pragma unroll
        for (int v = 0; v < 8; ++v) {
            float mx = s[0][v];
#pragma unroll
            for (int j = 1; j < 16; ++j) mx = fmaxf(mx, s[j][v]);
#pragma unroll
            for (int off = 1; off < 16; off <<= 1)
                mx = fmaxf(mx, __shfl_xor(mx, off, 32));
            float sum = 0.0f;
#pragma unroll
            for (int j = 0; j < 16; ++j) {
                float e = __expf(s[j][v] - mx);
                s[j][v] = e;
                sum += e;
            }
#pragma unroll
            for (int off = 1; off < 16; off <<= 1)
                sum += __shfl_xor(sum, off, 32);
            rinv[v] = 1.0f / sum;
        }

        // write normalized probabilities (f16) to this wave's strip
#pragma unroll
        for (int j = 0; j < 16; ++j) {
#pragma unroll
            for (int v = 0; v < 8; ++v) {
                const int m = v + (hi ? 8 : 0);
                sP[wave][m][j * 16 + ln16] = (_Float16)(s[j][v] * rinv[v]);
            }
        }

        // O = P @ V ; causality limits k-steps
        const int nks = (m0 + 16 + 31) / 32;
        f8v o[3];
        o[0] = zero8(); o[1] = zero8(); o[2] = zero8();
        for (int ks = 0; ks < nks; ++ks) {
            const int kb = ks * 32;
            F16frag pa;
            pa.h[0] = *(const h8v*)&sP[wave][ln16][kb + (hi ? 8 : 0)];
            pa.h[1] = *(const h8v*)&sP[wave][ln16][kb + 16 + (hi ? 8 : 0)];
#pragma unroll
            for (int t = 0; t < 3; ++t) {
                const int kb2 = kb + (hi ? 8 : 0);
                F16frag vf;
                vf.h[0] = *(const h8v*)&sVt[t * 16 + ln16][kb2];
                vf.h[1] = *(const h8v*)&sVt[t * 16 + ln16][kb2 + 16];
                o[t] = __builtin_amdgcn_wmma_f32_16x16x32_f16(
                    false, pa.v, false, vf.v, (short)0, o[t], false, false);
            }
        }

        // store output f32, lanes 0-15 contiguous in n
        float* ob = out + ((size_t)b * T_SZ + m0) * HS;
#pragma unroll
        for (int t = 0; t < 3; ++t) {
#pragma unroll
            for (int v = 0; v < 8; ++v) {
                const int m = v + (hi ? 8 : 0);
                ob[(size_t)m * HS + t * 16 + ln16] = o[t][v];
            }
        }
    }
}

extern "C" void kernel_launch(void* const* d_in, const int* in_sizes, int n_in,
                              void* d_out, int out_size, void* d_ws, size_t ws_size,
                              hipStream_t stream) {
    (void)in_sizes; (void)n_in; (void)out_size; (void)ws_size;
    const float* x  = (const float*)d_in[0];
    const float* Wq = (const float*)d_in[1];
    const float* Wk = (const float*)d_in[2];
    const float* Wv = (const float*)d_in[3];
    _Float16* wt = (_Float16*)d_ws;               // 3*48*384*2 = 110,592 B scratch
    float* out = (float*)d_out;

    const int total = 3 * HS * C_SZ;
    head_prep_w<<<(total + 255) / 256, 256, 0, stream>>>(Wq, Wk, Wv, wt);
    head_attn<<<B_SZ, 256, 0, stream>>>(x, wt, out);
}
